// FlashBiasAttention_27530740367698
// MI455X (gfx1250) — matrix operand order repaired and checked by reading, not compiled, not run on hardware
//
#include <hip/hip_runtime.h>
#include <hip/hip_bf16.h>

#define S_LEN 2048
#define EMB   1024
#define NH    16
#define HD    64

typedef __attribute__((ext_vector_type(16))) _Float16 v16h;
typedef __attribute__((ext_vector_type(8)))  _Float16 v8h;
typedef __attribute__((ext_vector_type(8)))  float    v8f;
typedef __attribute__((ext_vector_type(4)))  float    f32x4;
typedef __attribute__((ext_vector_type(4)))  unsigned int u32x4;

// D = A(16x32 f16) * B(32x16 f16) + C(16x16 f32), wave32 WMMA
__device__ __forceinline__ v8f wmma16x16x32(v16h a, v16h b, v8f c) {
  return __builtin_amdgcn_wmma_f32_16x16x32_f16(false, a, false, b, (short)0, c,
                                                false, false);
}

// --- CDNA5 async global->LDS copies (per-lane, tracked with ASYNCcnt) -------
__device__ __forceinline__ void async_ld16(void* lds, const void* g) {
  asm volatile("global_load_async_to_lds_b128 %0, %1, off"
               :: "v"((unsigned)(uintptr_t)lds), "v"(g) : "memory");
}
__device__ __forceinline__ void async_ld16_nt(void* lds, const void* g) {
  asm volatile("global_load_async_to_lds_b128 %0, %1, off th:TH_LOAD_NT"
               :: "v"((unsigned)(uintptr_t)lds), "v"(g) : "memory");
}
__device__ __forceinline__ void wait_async0() {
  asm volatile("s_wait_asynccnt 0x0" ::: "memory");
}

// A-fragment: tile stored row-major [m][k]; per ISA layout lanes 0-15 hold
// M=0..15 / K-halves {0..7, 16..23}, lanes 16-31 hold K-halves {8..15, 24..31}.
__device__ __forceinline__ v16h frag_a(const _Float16* __restrict__ base, int ld,
                                       int row0, int k0) {
  const int lane  = threadIdx.x & 31;
  const int mlo   = lane & 15;
  const int khalf = lane >> 4;
  const _Float16* p = base + (size_t)(row0 + mlo) * ld + k0 + khalf * 8;
  v16h r;
  ((v8h*)&r)[0] = *(const v8h*)(p);        // K = khalf*8 + 0..7
  ((v8h*)&r)[1] = *(const v8h*)(p + 16);   // K = 16 + khalf*8 + 0..7
  return r;
}

// B-fragment from TRANSPOSED storage [n][k] (column-major B): lane n reads
// K-halves {khalf*8+0..7, 16+khalf*8+0..7} (same order as frag_a).
__device__ __forceinline__ v16h frag_bt(const _Float16* __restrict__ base, int ld,
                                        int col0, int k0) {
  const int lane  = threadIdx.x & 31;
  const int nlo   = lane & 15;
  const int khalf = lane >> 4;
  const _Float16* p = base + (size_t)(col0 + nlo) * ld + k0 + khalf * 8;
  v16h r;
  ((v8h*)&r)[0] = *(const v8h*)(p);
  ((v8h*)&r)[1] = *(const v8h*)(p + 16);
  return r;
}

// ---------------------------------------------------------------------------
// Kernel 1: packed QKV projection.  Y = X @ W_chunk^T + b_chunk, emitted in
// f16 with attention-friendly layouts:
//   mode0: Qh[h][s][d]   mode1: Kh[h][s][d]   mode2: Vt[h][d][s]
// Block: 256 thr (8 waves), 64x64 output tile, K-step 32.
// (f32->f16 conversion forces the VGPR staging path here.)
// ---------------------------------------------------------------------------
__global__ void __launch_bounds__(256)
qkv_gemm_kernel(const float* __restrict__ query,
                const float* __restrict__ key_,
                const float* __restrict__ value,
                const float* __restrict__ w,     // [3E][E]
                const float* __restrict__ bvec,  // [3E]
                _Float16* __restrict__ Qh,
                _Float16* __restrict__ Kh,
                _Float16* __restrict__ Vt) {
  __shared__ _Float16 Xs[64 * 40];   // [m][k] row-major (A)
  __shared__ _Float16 Ws[64 * 40];   // [n][k] == transposed B (W natural layout)

  const int mode = blockIdx.z;
  const float* __restrict__ X  = (mode == 0) ? query : (mode == 1) ? key_ : value;
  const float* __restrict__ W  = w + (size_t)mode * EMB * EMB;
  const float* __restrict__ Bv = bvec + mode * EMB;

  const int mbase = blockIdx.y * 64;
  const int nbase = blockIdx.x * 64;
  const int t     = threadIdx.x;
  const int wave  = t >> 5;
  const int wm    = wave >> 1;   // 0..3: 16-row subtile
  const int wn    = wave & 1;    // 0..1: 32-col strip (2 x 16)
  const int srow  = t >> 2;      // staging: 64 rows x 4 parts x 8 elems
  const int spart = t & 3;

  v8f acc0 = {};
  v8f acc1 = {};

  for (int k0 = 0; k0 < EMB; k0 += 32) {
    __syncthreads();
    {
      const f32x4* gx = (const f32x4*)(X + (size_t)(mbase + srow) * EMB + k0 + spart * 8);
      f32x4 x0 = gx[0], x1 = gx[1];
      _Float16* sx = Xs + srow * 40 + spart * 8;
      sx[0] = (_Float16)x0.x; sx[1] = (_Float16)x0.y;
      sx[2] = (_Float16)x0.z; sx[3] = (_Float16)x0.w;
      sx[4] = (_Float16)x1.x; sx[5] = (_Float16)x1.y;
      sx[6] = (_Float16)x1.z; sx[7] = (_Float16)x1.w;

      const f32x4* gw = (const f32x4*)(W + (size_t)(nbase + srow) * EMB + k0 + spart * 8);
      f32x4 w0 = gw[0], w1 = gw[1];
      _Float16* sw = Ws + srow * 40 + spart * 8;
      sw[0] = (_Float16)w0.x; sw[1] = (_Float16)w0.y;
      sw[2] = (_Float16)w0.z; sw[3] = (_Float16)w0.w;
      sw[4] = (_Float16)w1.x; sw[5] = (_Float16)w1.y;
      sw[6] = (_Float16)w1.z; sw[7] = (_Float16)w1.w;

      // prefetch next K-tile into near caches (WGP scope)
      if (k0 + 32 < EMB) {
        __builtin_prefetch(X + (size_t)(mbase + srow) * EMB + k0 + 32 + spart * 8, 0, 3);
        __builtin_prefetch(W + (size_t)(nbase + srow) * EMB + k0 + 32 + spart * 8, 0, 3);
      }
    }
    __syncthreads();

    v16h a  = frag_a (Xs, 40, wm * 16, 0);
    v16h b0 = frag_bt(Ws, 40, wn * 32, 0);
    v16h b1 = frag_bt(Ws, 40, wn * 32 + 16, 0);
    acc0 = wmma16x16x32(a, b0, acc0);
    acc1 = wmma16x16x32(a, b1, acc1);
  }

  const int lane = t & 31;
  const int half = lane >> 4;
  const int nlo  = lane & 15;
#pragma unroll
  for (int sub = 0; sub < 2; ++sub) {
    v8f acc = sub ? acc1 : acc0;
    const int col = nbase + wn * 32 + sub * 16 + nlo;
    const float bias_c = Bv[col];
    const int h = col >> 6;
    const int d = col & 63;
#pragma unroll
    for (int r = 0; r < 8; ++r) {
      const int row = mbase + wm * 16 + r + half * 8;   // C layout: M = r + 8*half
      const float val = acc[r] + bias_c;
      if (mode == 0)      Qh[((size_t)h * S_LEN + row) * HD + d] = (_Float16)val;
      else if (mode == 1) Kh[((size_t)h * S_LEN + row) * HD + d] = (_Float16)val;
      else                Vt[((size_t)h * HD + d) * S_LEN + row] = (_Float16)val;
    }
  }
}

// ---------------------------------------------------------------------------
// Kernel 2: flash attention with additive bias.
// Block: 128 thr (4 waves). Each wave owns one 16-row Q tile; the block
// covers 64 query rows of one head and streams 32-key chunks of K/V/bias
// through LDS using async global->LDS copies (ASYNCcnt).
// ---------------------------------------------------------------------------
__global__ void __launch_bounds__(128)
attn_kernel(const _Float16* __restrict__ Qh,
            const _Float16* __restrict__ Kh,
            const _Float16* __restrict__ Vt,
            const float* __restrict__ bias,   // [H][S][S]
            _Float16* __restrict__ CTX) {     // [S][E] f16
  __shared__ _Float16 Ks[32 * 80];      // [sk][d]  (transposed-B for QK^T)
  __shared__ _Float16 Vs[64 * 32];      // [d][sk]  (transposed-B for P@V)
  __shared__ float    Bs[64 * 36];      // [m][sk] bias tile
  __shared__ _Float16 Ps[4][16 * 32];   // per-wave P tile (A layout source)

  const int h     = blockIdx.y;
  const int qbase = blockIdx.x * 64;
  const int t     = threadIdx.x;
  const int wave  = t >> 5;
  const int lane  = t & 31;
  const int mlo   = lane & 15;
  const int half  = lane >> 4;

  const int qrow0 = qbase + wave * 16;
  const _Float16* qptr = Qh + ((size_t)h * S_LEN + qrow0) * HD;
  const v16h q0 = frag_a(qptr, HD, 0, 0);    // d = 0..31
  const v16h q1 = frag_a(qptr, HD, 0, 32);   // d = 32..63

  float mrow[8], lrow[8];
  v8f c0 = {}, c1 = {}, c2 = {}, c3 = {};
#pragma unroll
  for (int r = 0; r < 8; ++r) { mrow[r] = -1e30f; lrow[r] = 0.0f; }

  const float scale = 0.125f;             // 1/sqrt(64)
  const int krow = t >> 2, kpart = t & 3; // K staging: 32 rows x 4 x 16 halfs
  const int vrow = t >> 1, vpart = t & 1; // V/bias staging: 64 rows x 2 x 16

  for (int kb = 0; kb < S_LEN; kb += 32) {
    __syncthreads();   // previous tile fully consumed
    {  // K chunk: Kh[h][kb+row][0..64] -> Ks[row][d], ld=80 halfs
      const _Float16* g = Kh + ((size_t)h * S_LEN + kb + krow) * HD + kpart * 16;
      _Float16* sd = Ks + krow * 80 + kpart * 16;
      async_ld16(sd,     g);
      async_ld16(sd + 8, g + 8);
    }
    {  // V chunk: Vt[h][d][kb..kb+32] -> Vs[d][sk], ld=32 halfs
      const _Float16* g = Vt + ((size_t)h * HD + vrow) * S_LEN + kb + vpart * 16;
      _Float16* sd = Vs + vrow * 32 + vpart * 16;
      async_ld16(sd,     g);
      async_ld16(sd + 8, g + 8);
    }
    {  // bias tile (single-use 268MB stream: non-temporal)
      const float* g = bias + (size_t)h * S_LEN * S_LEN +
                       (size_t)(qbase + vrow) * S_LEN + kb + vpart * 16;
      float* sd = Bs + vrow * 36 + vpart * 16;
      async_ld16_nt(sd,      g);
      async_ld16_nt(sd + 4,  g + 4);
      async_ld16_nt(sd + 8,  g + 8);
      async_ld16_nt(sd + 12, g + 12);
    }
    wait_async0();     // this wave's async copies landed in LDS
    __syncthreads();   // all waves' copies visible

    // scores: 16(q) x 32(k), two 16x16 N-subtiles, K-contraction = 64 (2 wmma)
    v8f s0 = {}, s1 = {};
    s0 = wmma16x16x32(q0, frag_bt(Ks, 80, 0, 0),   s0);
    s0 = wmma16x16x32(q1, frag_bt(Ks, 80, 0, 32),  s0);
    s1 = wmma16x16x32(q0, frag_bt(Ks, 80, 16, 0),  s1);
    s1 = wmma16x16x32(q1, frag_bt(Ks, 80, 16, 32), s1);

#pragma unroll
    for (int r = 0; r < 8; ++r) {
      const int m = wave * 16 + r + half * 8;
      s0[r] = s0[r] * scale + Bs[m * 36 + mlo];
      s1[r] = s1[r] * scale + Bs[m * 36 + 16 + mlo];
    }

    // online softmax; row reductions stay inside each 16-lane half (wave32)
#pragma unroll
    for (int r = 0; r < 8; ++r) {
      float v = fmaxf(s0[r], s1[r]);
      v = fmaxf(v, __shfl_xor(v, 1, 32));
      v = fmaxf(v, __shfl_xor(v, 2, 32));
      v = fmaxf(v, __shfl_xor(v, 4, 32));
      v = fmaxf(v, __shfl_xor(v, 8, 32));
      const float mn   = fmaxf(mrow[r], v);
      const float corr = __expf(mrow[r] - mn);
      mrow[r] = mn;
      const float p0 = __expf(s0[r] - mn);
      const float p1 = __expf(s1[r] - mn);
      float ps = p0 + p1;
      ps += __shfl_xor(ps, 1, 32);
      ps += __shfl_xor(ps, 2, 32);
      ps += __shfl_xor(ps, 4, 32);
      ps += __shfl_xor(ps, 8, 32);
      lrow[r] = lrow[r] * corr + ps;
      c0[r] *= corr; c1[r] *= corr; c2[r] *= corr; c3[r] *= corr;
      const int m = r + half * 8;
      Ps[wave][m * 32 + mlo]      = (_Float16)p0;   // C-layout -> A-layout via LDS
      Ps[wave][m * 32 + 16 + mlo] = (_Float16)p1;
    }

    asm volatile("s_wait_dscnt 0" ::: "memory");    // P stores visible to frag read

    const v16h p = frag_a(&Ps[wave][0], 32, 0, 0);
    c0 = wmma16x16x32(p, frag_bt(Vs, 32, 0, 0),  c0);
    c1 = wmma16x16x32(p, frag_bt(Vs, 32, 16, 0), c1);
    c2 = wmma16x16x32(p, frag_bt(Vs, 32, 32, 0), c2);
    c3 = wmma16x16x32(p, frag_bt(Vs, 32, 48, 0), c3);
  }

#pragma unroll
  for (int r = 0; r < 8; ++r) {
    const int row = qrow0 + r + half * 8;
    const float inv = 1.0f / lrow[r];
    _Float16* o = CTX + (size_t)row * EMB + h * HD;
    o[mlo]      = (_Float16)(c0[r] * inv);
    o[16 + mlo] = (_Float16)(c1[r] * inv);
    o[32 + mlo] = (_Float16)(c2[r] * inv);
    o[48 + mlo] = (_Float16)(c3[r] * inv);
  }
}

// ---------------------------------------------------------------------------
// Kernel 3: output projection.  out = CTX @ out_w^T + out_b  (f32 output)
// CTX is already f16 -> its staging uses async global->LDS copies.
// ---------------------------------------------------------------------------
__global__ void __launch_bounds__(256)
out_gemm_kernel(const _Float16* __restrict__ CTX,
                const float* __restrict__ W,    // [E][E] natural = transposed-B
                const float* __restrict__ Bv,   // [E]
                float* __restrict__ out) {
  __shared__ _Float16 Xs[64 * 40];
  __shared__ _Float16 Ws[64 * 40];

  const int mbase = blockIdx.y * 64;
  const int nbase = blockIdx.x * 64;
  const int t     = threadIdx.x;
  const int wave  = t >> 5;
  const int wm    = wave >> 1;
  const int wn    = wave & 1;
  const int srow  = t >> 2;
  const int spart = t & 3;

  v8f acc0 = {}, acc1 = {};

  for (int k0 = 0; k0 < EMB; k0 += 32) {
    __syncthreads();
    {
      // f16 tile: async copy straight into LDS
      async_ld16(Xs + srow * 40 + spart * 8,
                 CTX + (size_t)(mbase + srow) * EMB + k0 + spart * 8);

      const f32x4* gw = (const f32x4*)(W + (size_t)(nbase + srow) * EMB + k0 + spart * 8);
      f32x4 w0 = gw[0], w1 = gw[1];
      _Float16* sw = Ws + srow * 40 + spart * 8;
      sw[0] = (_Float16)w0.x; sw[1] = (_Float16)w0.y;
      sw[2] = (_Float16)w0.z; sw[3] = (_Float16)w0.w;
      sw[4] = (_Float16)w1.x; sw[5] = (_Float16)w1.y;
      sw[6] = (_Float16)w1.z; sw[7] = (_Float16)w1.w;

      if (k0 + 32 < EMB) {
        __builtin_prefetch(W + (size_t)(nbase + srow) * EMB + k0 + 32 + spart * 8, 0, 3);
      }
    }
    wait_async0();
    __syncthreads();

    v16h a  = frag_a (Xs, 40, wm * 16, 0);
    v16h b0 = frag_bt(Ws, 40, wn * 32, 0);
    v16h b1 = frag_bt(Ws, 40, wn * 32 + 16, 0);
    acc0 = wmma16x16x32(a, b0, acc0);
    acc1 = wmma16x16x32(a, b1, acc1);
  }

  const int lane = t & 31;
  const int half = lane >> 4;
  const int nlo  = lane & 15;
#pragma unroll
  for (int sub = 0; sub < 2; ++sub) {
    v8f acc = sub ? acc1 : acc0;
    const int col = nbase + wn * 32 + sub * 16 + nlo;
    const float bias_c = Bv[col];
#pragma unroll
    for (int r = 0; r < 8; ++r) {
      const int row = mbase + wm * 16 + r + half * 8;
      out[(size_t)row * EMB + col] = acc[r] + bias_c;
    }
  }
}

// ---------------------------------------------------------------------------
extern "C" void kernel_launch(void* const* d_in, const int* in_sizes, int n_in,
                              void* d_out, int out_size, void* d_ws, size_t ws_size,
                              hipStream_t stream) {
  (void)in_sizes; (void)n_in; (void)out_size; (void)ws_size;

  const float* query          = (const float*)d_in[0];
  const float* key_           = (const float*)d_in[1];
  const float* value          = (const float*)d_in[2];
  const float* attn_bias      = (const float*)d_in[3];
  const float* in_proj_weight = (const float*)d_in[4];
  const float* in_proj_bias   = (const float*)d_in[5];
  const float* out_w          = (const float*)d_in[6];
  const float* out_b          = (const float*)d_in[7];
  float* out = (float*)d_out;

  // Workspace layout (f16): Qh | Kh | Vt | CTX  = 4 x 4 MB = 16 MB
  _Float16* wsh = (_Float16*)d_ws;
  const size_t per = (size_t)NH * S_LEN * HD;  // 2,097,152 elements
  _Float16* Qh  = wsh;
  _Float16* Kh  = wsh + per;
  _Float16* Vt  = wsh + 2 * per;
  _Float16* CTX = wsh + 3 * per;

  dim3 g1(EMB / 64, S_LEN / 64, 3);
  qkv_gemm_kernel<<<g1, dim3(256), 0, stream>>>(query, key_, value,
                                                in_proj_weight, in_proj_bias,
                                                Qh, Kh, Vt);

  dim3 g2(S_LEN / 64, NH);
  attn_kernel<<<g2, dim3(128), 0, stream>>>(Qh, Kh, Vt, attn_bias, CTX);

  dim3 g3(EMB / 64, S_LEN / 64);
  out_gemm_kernel<<<g3, dim3(256), 0, stream>>>(CTX, out_w, out_b, out);
}
